// ProGenBase_46806553592398
// MI455X (gfx1250) — matrix operand order repaired and checked by reading, not compiled, not run on hardware
//
#include <hip/hip_runtime.h>

typedef __bf16 bf16;
typedef __attribute__((ext_vector_type(8)))  bf16  v8bf;
typedef __attribute__((ext_vector_type(16))) bf16  v16bf;
typedef __attribute__((ext_vector_type(8)))  float v8f;

#define N_TOK 2048
#define DIM   1024
#define DFF   4096
#define HEADS 8
#define DHEAD 64
#define WSZ   512
#define NWIN  4
#define VOCAB 32

// ---------------------------------------------------------------- helpers

__device__ __forceinline__ float gelu_f(float x) {
  float x3 = x * x * x;
  return 0.5f * x * (1.0f + tanhf(0.7978845608028654f * (x + 0.044715f * x3)));
}

// A fragment 16x32 bf16: lane l -> row m0+(l&15); lanes 0-15 hold K 0..7 &
// 16..23, lanes 16-31 hold K 8..15 & 24..31 (ISA 7.12.2). Two 16B loads.
__device__ __forceinline__ v16bf ld_afrag(const bf16* p) {
  v8bf lo = *(const v8bf*)p;
  v8bf hi = *(const v8bf*)(p + 16);
  return __builtin_shufflevector(lo, hi, 0,1,2,3,4,5,6,7,8,9,10,11,12,13,14,15);
}
// B fragment 32x16 bf16 from TRANSPOSED weight Bt[N][K]: lane l -> col
// n0+(l&15), lanes 0-15 hold K 0..7 & 16..23, lanes 16-31 hold K 8..15 & 24..31 (two 16B loads).
__device__ __forceinline__ v16bf ld_bfrag(const bf16* p) {
  v8bf lo = *(const v8bf*)p;
  v8bf hi = *(const v8bf*)(p + 16);
  return __builtin_shufflevector(lo, hi, 0,1,2,3,4,5,6,7,8,9,10,11,12,13,14,15);
}

// ---------------------------------------------------------------- GEMM
// C[m,n] = alpha * sum_k A[m,k]*Bt[n,k]  (+bias[n]) (+rbias[m])
//          (sim-mask) (gelu) (+resid[m,n]) ; writes fp32 and/or bf16.
// Each wave: 32x32 C block via 4x v_wmma_f32_16x16x32_bf16 per k-step.
__global__ __launch_bounds__(256) void gemm_bf16_k(
    const bf16* __restrict__ A, int lda, long long sA,
    const bf16* __restrict__ B, int ldb, long long sB,
    float* Cf, bf16* Cb, int ldc, long long sC,
    const float* __restrict__ bias, const float* __restrict__ rbias,
    const float* resid, int ldr,
    int M, int N, int K, float alpha, int act, int simmask, int cmode)
{
  const int lane  = threadIdx.x & 31;
  const int wave  = threadIdx.x >> 5;
  const int batch = blockIdx.y;
  const int mt = M >> 5, nt = N >> 5;
  int tile = blockIdx.x * 8 + wave;
  if (tile >= mt * nt) return;              // wave-uniform: EXEC stays full
  const int m0 = (tile % mt) << 5;
  const int n0 = (tile / mt) << 5;

  const bf16* Ab = A + (long long)batch * sA;
  const bf16* Bb = B + (long long)batch * sB;
  long long coff;
  if (cmode == 1) {                         // attn-out scatter: [w*512, h*64]
    int h = batch >> 2, w = batch & 3;
    coff = (long long)(w * WSZ) * ldc + h * DHEAD;
  } else {
    coff = (long long)batch * sC;
  }

  const int r16  = lane & 15;
  const int half = lane >> 4;
  const bf16* pa0 = Ab + (long long)(m0 + r16) * lda + half * 8;
  const bf16* pa1 = pa0 + (long long)16 * lda;
  const bf16* pb0 = Bb + (long long)(n0 + r16) * ldb + half * 8;
  const bf16* pb1 = pb0 + (long long)16 * ldb;

  v8f c00 = {}, c01 = {}, c10 = {}, c11 = {};
  for (int k0 = 0; k0 < K; k0 += 32) {
    v16bf a0 = ld_afrag(pa0 + k0);
    v16bf a1 = ld_afrag(pa1 + k0);
    v16bf b0 = ld_bfrag(pb0 + k0);
    v16bf b1 = ld_bfrag(pb1 + k0);
    c00 = __builtin_amdgcn_wmma_f32_16x16x32_bf16(false, a0, false, b0, (short)0, c00, false, false);
    c01 = __builtin_amdgcn_wmma_f32_16x16x32_bf16(false, a0, false, b1, (short)0, c01, false, false);
    c10 = __builtin_amdgcn_wmma_f32_16x16x32_bf16(false, a1, false, b0, (short)0, c10, false, false);
    c11 = __builtin_amdgcn_wmma_f32_16x16x32_bf16(false, a1, false, b1, (short)0, c11, false, false);
  }

  #pragma unroll
  for (int t = 0; t < 4; ++t) {
    v8f c = (t == 0) ? c00 : (t == 1) ? c01 : (t == 2) ? c10 : c11;
    int mb = m0 + ((t >> 1) << 4) + (half << 3);   // C: VGPR r -> M=r+8*(l>>4)
    int n  = n0 + ((t & 1) << 4) + r16;            //    N = l&15
    float bn = bias ? bias[n] : 0.0f;
    #pragma unroll
    for (int r = 0; r < 8; ++r) {
      int m = mb + r;
      float v = c[r] * alpha + bn;
      if (rbias) v += rbias[m];
      if (simmask && n > m + WSZ) v = -1e10f;      // tril(ones(512,1024), 512)
      if (act) v = gelu_f(v);
      if (resid) v += resid[(long long)m * ldr + n];
      long long o = coff + (long long)m * ldc + n;
      if (Cf) Cf[o] = v;
      if (Cb) Cb[o] = (bf16)v;
    }
  }
}

// ---------------------------------------------------------------- aux kernels

__global__ void embed_k(const int* __restrict__ tok, const float* __restrict__ emb,
                        float* __restrict__ x) {
  long long i = (long long)blockIdx.x * 256 + threadIdx.x;
  if (i >= (long long)N_TOK * DIM) return;
  int c = i & (DIM - 1);
  int n = i >> 10;
  x[i] = emb[(long long)tok[n] * DIM + c];
}

// W[K][N] fp32 -> Wt[N][K] bf16 (transposed for B-fragment contiguity)
__global__ void convT_k(const float* __restrict__ W, bf16* __restrict__ Wt,
                        int K, int N) {
  long long i = (long long)blockIdx.x * 256 + threadIdx.x;
  if (i >= (long long)K * N) return;
  int n = (int)(i % N);
  int k = (int)(i / N);
  Wt[(long long)n * K + k] = (bf16)W[i];
}

// spatial-gate weight: bf16 with causal (n<=m) mask folded in
__global__ void convmask_k(const float* __restrict__ S, bf16* __restrict__ Sb, int n) {
  long long i = (long long)blockIdx.x * 256 + threadIdx.x;
  if (i >= (long long)n * n) return;
  int col = (int)(i % n), row = (int)(i / n);
  Sb[i] = (col <= row) ? (bf16)S[i] : (bf16)0.0f;
}

// LayerNorm(row of `dim`) * g -> bf16, optionally transposed output
__global__ __launch_bounds__(256) void ln_k(const float* __restrict__ x, int xld,
                                            const float* __restrict__ g,
                                            bf16* __restrict__ out, int dim, int tld) {
  int row = blockIdx.x, tid = threadIdx.x;
  const float* xr = x + (long long)row * xld;
  __shared__ float s1[256], s2[256];
  float a = 0.f, b = 0.f;
  for (int c = tid; c < dim; c += 256) { float v = xr[c]; a += v; b += v * v; }
  s1[tid] = a; s2[tid] = b; __syncthreads();
  for (int s = 128; s > 0; s >>= 1) {
    if (tid < s) { s1[tid] += s1[tid + s]; s2[tid] += s2[tid + s]; }
    __syncthreads();
  }
  float mean = s1[0] / dim;
  float inv  = rsqrtf(s2[0] / dim - mean * mean + 1e-5f);
  for (int c = tid; c < dim; c += 256) {
    float y = (xr[c] - mean) * inv * g[c];
    if (tld) out[(long long)c * tld + row] = (bf16)y;
    else     out[(long long)row * dim + c] = (bf16)y;
  }
}

// qkv bf16 [2048][1536] -> qb[h][n][d], lookback kb[b][j][d], vT vtb[b][d][j]
__global__ void prep_attn_k(const bf16* __restrict__ qkv, bf16* __restrict__ qb,
                            bf16* __restrict__ kb, bf16* __restrict__ vtb) {
  long long i = (long long)blockIdx.x * 256 + threadIdx.x;
  const long long NQ = (long long)HEADS * N_TOK * DHEAD;      // 1M
  if (i < NQ) {
    int d = i & 63;
    int n = (int)((i >> 6) & (N_TOK - 1));
    int h = (int)(i >> 17);
    qb[i] = qkv[(long long)n * 1536 + h * DHEAD + d];
    return;
  }
  long long j = i - NQ;                                        // 32*1024*64
  if (j >= (long long)32 * 1024 * 64) return;
  int d  = (int)(j & 63);
  int jj = (int)((j >> 6) & 1023);
  int b  = (int)(j >> 16);
  int h = b >> 2, w = b & 3;
  bf16 kv = (bf16)0.0f, vv = (bf16)0.0f;
  int src; bool valid;
  if (jj < WSZ) { src = (w - 1) * WSZ + jj; valid = (w > 0); }   // prev window
  else          { src = w * WSZ + (jj - WSZ); valid = true; }    // cur window
  if (valid) {
    kv = qkv[(long long)src * 1536 + 512  + h * DHEAD + d];
    vv = qkv[(long long)src * 1536 + 1024 + h * DHEAD + d];
  }
  kb [(long long)b * 65536 + (long long)jj * 64 + d] = kv;
  vtb[(long long)b * 65536 + (long long)d * 1024 + jj] = vv;
}

// row softmax over 1024 (sim fp32 -> attn bf16)
__global__ __launch_bounds__(256) void softmax_k(const float* __restrict__ sim,
                                                 bf16* __restrict__ attn) {
  long long row = blockIdx.x;
  int tid = threadIdx.x;
  const float* sr = sim + row * 1024;
  __shared__ float sh[256];
  float mx = -3.4e38f;
  #pragma unroll
  for (int k = 0; k < 4; ++k) mx = fmaxf(mx, sr[tid + k * 256]);
  sh[tid] = mx; __syncthreads();
  for (int s = 128; s > 0; s >>= 1) {
    if (tid < s) sh[tid] = fmaxf(sh[tid], sh[tid + s]);
    __syncthreads();
  }
  mx = sh[0]; __syncthreads();
  float e[4]; float sum = 0.f;
  #pragma unroll
  for (int k = 0; k < 4; ++k) { e[k] = __expf(sr[tid + k * 256] - mx); sum += e[k]; }
  sh[tid] = sum; __syncthreads();
  for (int s = 128; s > 0; s >>= 1) {
    if (tid < s) sh[tid] += sh[tid + s];
    __syncthreads();
  }
  float inv = 1.0f / sh[0];
  #pragma unroll
  for (int k = 0; k < 4; ++k) attn[row * 1024 + tid + k * 256] = (bf16)(e[k] * inv);
}

// u = bf16( xg[m,d] * gate_out[m,d] ) ; xg = first half of h[2048][4096]
__global__ void mulgate_k(const float* __restrict__ h, const float* __restrict__ go,
                          bf16* __restrict__ u) {
  long long i = (long long)blockIdx.x * 256 + threadIdx.x;
  if (i >= (long long)N_TOK * 2048) return;
  int d = (int)(i & 2047);
  long long m = i >> 11;
  u[i] = (bf16)(h[m * DFF + d] * go[i]);
}

// ---------------------------------------------------------------- driver

extern "C" void kernel_launch(void* const* d_in, const int* in_sizes, int n_in,
                              void* d_out, int out_size, void* d_ws, size_t ws_size,
                              hipStream_t stream) {
  (void)in_sizes; (void)n_in; (void)out_size; (void)ws_size;
  const int*   tokens  = (const int*)  d_in[0];
  const float* embed   = (const float*)d_in[1];
  const float* attn_ln = (const float*)d_in[2];
  const float* attn_qkv= (const float*)d_in[3];
  const float* attn_ow = (const float*)d_in[4];
  const float* attn_ob = (const float*)d_in[5];
  const float* ff_ln   = (const float*)d_in[6];
  const float* ff_w1   = (const float*)d_in[7];
  const float* ff_b1   = (const float*)d_in[8];
  const float* ff_w2   = (const float*)d_in[9];
  const float* ff_b2   = (const float*)d_in[10];
  const float* g_ln    = (const float*)d_in[11];
  const float* g_win   = (const float*)d_in[12];
  const float* g_bin   = (const float*)d_in[13];
  const float* g_sln   = (const float*)d_in[14];
  const float* g_sw    = (const float*)d_in[15];
  const float* g_sb    = (const float*)d_in[16];
  const float* g_sow   = (const float*)d_in[17];
  const float* g_sob   = (const float*)d_in[18];
  const float* g_pow   = (const float*)d_in[19];
  const float* g_pob   = (const float*)d_in[20];
  const float* fin_ln  = (const float*)d_in[21];
  const float* head_w  = (const float*)d_in[22];
  const float* head_b  = (const float*)d_in[23];

  // ---- workspace layout (phase-overlapped) ----
  char* ws = (char*)d_ws;
  size_t off = 0;
  auto bump = [&](size_t bytes) -> char* {
    char* p = ws + off;
    off += (bytes + 255) & ~(size_t)255;
    return p;
  };
  float* x    = (float*)bump((size_t)N_TOK * DIM * 4);        // residual stream
  bf16*  xn   = (bf16*) bump((size_t)N_TOK * DIM * 2);        // LN'd input
  bf16*  wbuf = (bf16*) bump((size_t)DFF * DIM * 2);          // transposed weight
  size_t phase = off;

  // attention/FF phase
  off = phase;
  bf16*  qkvb = (bf16*) bump((size_t)N_TOK * 1536 * 2);
  bf16*  qb   = (bf16*) bump((size_t)HEADS * N_TOK * DHEAD * 2);
  bf16*  kb   = (bf16*) bump((size_t)32 * 1024 * 64 * 2);
  bf16*  vtb  = (bf16*) bump((size_t)32 * 1024 * 64 * 2);
  float* sim  = (float*)bump((size_t)32 * 512 * 1024 * 4);
  bf16*  attn = (bf16*) bump((size_t)32 * 512 * 1024 * 2);
  bf16*  aob  = (bf16*) bump((size_t)N_TOK * 512 * 2);
  bf16*  h1b  = (bf16*) bump((size_t)N_TOK * DFF * 2);

  // gMLP phase (overlaps attention phase scratch)
  off = phase;
  float* hf    = (float*)bump((size_t)N_TOK * DFF * 4);
  bf16*  gateT = (bf16*) bump((size_t)2048 * N_TOK * 2);
  float* gateo = (float*)bump((size_t)N_TOK * 2048 * 4);
  bf16*  ub    = (bf16*) bump((size_t)N_TOK * 2048 * 2);
  bf16*  u2b   = (bf16*) bump((size_t)N_TOK * 2048 * 2);
  bf16*  swb   = (bf16*) bump((size_t)N_TOK * N_TOK * 2);

  auto blocks = [](long long n) { return (unsigned)((n + 255) / 256); };

  auto gemm = [&](const bf16* A, int lda, long long sA,
                  const bf16* B, int ldb, long long sB,
                  float* Cf, bf16* Cb, int ldc, long long sC,
                  const float* bias, const float* rbias,
                  const float* resid, int ldr,
                  int M, int N, int K, float alpha, int act,
                  int simmask, int cmode, int batch) {
    int tiles = (M / 32) * (N / 32);
    dim3 g((unsigned)((tiles + 7) / 8), (unsigned)batch);
    gemm_bf16_k<<<g, 256, 0, stream>>>(A, lda, sA, B, ldb, sB, Cf, Cb, ldc, sC,
                                       bias, rbias, resid, ldr, M, N, K,
                                       alpha, act, simmask, cmode);
  };

  // ---- embedding ----
  embed_k<<<blocks((long long)N_TOK * DIM), 256, 0, stream>>>(tokens, embed, x);

  // ---- 6 attention + FF blocks ----
  for (int d = 0; d < 6; ++d) {
    // local attention
    ln_k<<<N_TOK, 256, 0, stream>>>(x, DIM, attn_ln + (size_t)d * DIM, xn, DIM, 0);
    convT_k<<<blocks((long long)DIM * 1536), 256, 0, stream>>>(
        attn_qkv + (size_t)d * DIM * 1536, wbuf, DIM, 1536);
    gemm(xn, DIM, 0, wbuf, DIM, 0, nullptr, qkvb, 1536, 0,
         nullptr, nullptr, nullptr, 0, N_TOK, 1536, DIM, 1.f, 0, 0, 0, 1);
    prep_attn_k<<<blocks((long long)(1048576 + 2097152)), 256, 0, stream>>>(
        qkvb, qb, kb, vtb);
    // sim = q @ k^T * d^-0.5, causal-masked   (batch = h*4+w)
    gemm(qb, DHEAD, 512 * 64, kb, DHEAD, 1024 * 64, sim, nullptr, 1024, 512 * 1024,
         nullptr, nullptr, nullptr, 0, WSZ, 1024, DHEAD, 0.125f, 0, 1, 0, 32);
    softmax_k<<<32 * 512, 256, 0, stream>>>(sim, attn);
    // out = attn @ v  -> scatter to [n, h*64+d]
    gemm(attn, 1024, (long long)512 * 1024, vtb, 1024, 64 * 1024,
         nullptr, aob, 512, 0, nullptr, nullptr, nullptr, 0,
         WSZ, DHEAD, 1024, 1.f, 0, 0, 1, 32);
    // out proj + residual (in place)
    convT_k<<<blocks((long long)512 * DIM), 256, 0, stream>>>(
        attn_ow + (size_t)d * 512 * DIM, wbuf, 512, DIM);
    gemm(aob, 512, 0, wbuf, 512, 0, x, nullptr, DIM, 0,
         attn_ob + (size_t)d * DIM, nullptr, x, DIM,
         N_TOK, DIM, 512, 1.f, 0, 0, 0, 1);

    // feedforward
    ln_k<<<N_TOK, 256, 0, stream>>>(x, DIM, ff_ln + (size_t)d * DIM, xn, DIM, 0);
    convT_k<<<blocks((long long)DIM * DFF), 256, 0, stream>>>(
        ff_w1 + (size_t)d * DIM * DFF, wbuf, DIM, DFF);
    gemm(xn, DIM, 0, wbuf, DIM, 0, nullptr, h1b, DFF, 0,
         ff_b1 + (size_t)d * DFF, nullptr, nullptr, 0,
         N_TOK, DFF, DIM, 1.f, 1 /*gelu*/, 0, 0, 1);
    convT_k<<<blocks((long long)DFF * DIM), 256, 0, stream>>>(
        ff_w2 + (size_t)d * DFF * DIM, wbuf, DFF, DIM);
    gemm(h1b, DFF, 0, wbuf, DFF, 0, x, nullptr, DIM, 0,
         ff_b2 + (size_t)d * DIM, nullptr, x, DIM,
         N_TOK, DIM, DFF, 1.f, 0, 0, 0, 1);
  }

  // ---- 2 gMLP blocks ----
  for (int g = 0; g < 2; ++g) {
    ln_k<<<N_TOK, 256, 0, stream>>>(x, DIM, g_ln + (size_t)g * DIM, xn, DIM, 0);
    convT_k<<<blocks((long long)DIM * DFF), 256, 0, stream>>>(
        g_win + (size_t)g * DIM * DFF, wbuf, DIM, DFF);
    gemm(xn, DIM, 0, wbuf, DIM, 0, hf, nullptr, DFF, 0,
         g_bin + (size_t)g * DFF, nullptr, nullptr, 0,
         N_TOK, DFF, DIM, 1.f, 1 /*gelu*/, 0, 0, 1);
    // gate = LN(h[:,2048:]) -> transposed bf16 [d][n]
    ln_k<<<N_TOK, 256, 0, stream>>>(hf + 2048, DFF, g_sln + (size_t)g * 2048,
                                    gateT, 2048, N_TOK);
    // causal spatial gate: gateo[m,d] = sum_{n<=m} sw[m,n]*gate[n,d] + sb[m]
    convmask_k<<<blocks((long long)N_TOK * N_TOK), 256, 0, stream>>>(
        g_sw + (size_t)g * N_TOK * N_TOK, swb, N_TOK);
    gemm(swb, N_TOK, 0, gateT, N_TOK, 0, gateo, nullptr, 2048, 0,
         nullptr, g_sb + (size_t)g * N_TOK, nullptr, 0,
         N_TOK, 2048, N_TOK, 1.f, 0, 0, 0, 1);
    mulgate_k<<<blocks((long long)N_TOK * 2048), 256, 0, stream>>>(hf, gateo, ub);
    convT_k<<<blocks((long long)2048 * 2048), 256, 0, stream>>>(
        g_sow + (size_t)g * 2048 * 2048, wbuf, 2048, 2048);
    gemm(ub, 2048, 0, wbuf, 2048, 0, nullptr, u2b, 2048, 0,
         g_sob + (size_t)g * 2048, nullptr, nullptr, 0,
         N_TOK, 2048, 2048, 1.f, 0, 0, 0, 1);
    convT_k<<<blocks((long long)2048 * DIM), 256, 0, stream>>>(
        g_pow + (size_t)g * 2048 * DIM, wbuf, 2048, DIM);
    gemm(u2b, 2048, 0, wbuf, 2048, 0, x, nullptr, DIM, 0,
         g_pob + (size_t)g * DIM, nullptr, x, DIM,
         N_TOK, DIM, 2048, 1.f, 0, 0, 0, 1);
  }

  // ---- final LN + head ----
  ln_k<<<N_TOK, 256, 0, stream>>>(x, DIM, fin_ln, xn, DIM, 0);
  convT_k<<<blocks((long long)DIM * VOCAB), 256, 0, stream>>>(head_w, wbuf, DIM, VOCAB);
  gemm(xn, DIM, 0, wbuf, DIM, 0, (float*)d_out, nullptr, VOCAB, 0,
       head_b, nullptr, nullptr, 0, N_TOK, VOCAB, DIM, 1.f, 0, 0, 0, 1);
}
